// Discriminator_10986526343448
// MI455X (gfx1250) — matrix operand order repaired and checked by reading, not compiled, not run on hardware
//
#include <hip/hip_runtime.h>
#include <hip/hip_bf16.h>

typedef unsigned short u16;
typedef __attribute__((ext_vector_type(16))) __bf16 v16bf;
typedef __attribute__((ext_vector_type(8)))  float  v8f;

#define Bsz 128
#define Tsz 256
#define Hd  512
#define Din 1024
#define G3  1536
#define KC  128          // K-chunk staged in LDS per GEMM block

// ---------- helpers ----------

static __device__ __forceinline__ u16 f2bf(float f) {
  unsigned x = __float_as_uint(f);
  unsigned r = x + 0x7FFFu + ((x >> 16) & 1u);   // round-to-nearest-even
  return (u16)(r >> 16);
}

union V16U { uint4 q[2]; v16bf v; };

// A fragment (16x32 bf16): lanes 0-15 -> M=lane, K=k0+0..7 / k0+16..23;
// lanes 16-31 -> M=lane-16, K=k0+8..15 / k0+24..31.
static __device__ __forceinline__ v16bf load_a_frag(const u16* __restrict__ rowbase,
                                                    int stride, int lane, int k0) {
  const int m  = lane & 15;
  const int kk = k0 + ((lane >> 4) << 3);
  const u16* p = rowbase + (long)m * stride + kk;
  V16U u;
  u.q[0] = *(const uint4*)p;
  u.q[1] = *(const uint4*)(p + 16);
  return u.v;
}

// B fragment (32x16 bf16) from row-major W[N,K]: lanes 0-15: N=lane, K=k0+0..7 / k0+16..23;
// lanes 16-31: N=lane-16, K=k0+8..15 / k0+24..31.
static __device__ __forceinline__ v16bf load_b_frag(const u16* __restrict__ colbase,
                                                    int stride, int lane, int k0) {
  const int n  = lane & 15;
  const int kk = k0 + ((lane >> 4) << 3);
  const u16* p = colbase + (long)n * stride + kk;
  V16U u;
  u.q[0] = *(const uint4*)p;
  u.q[1] = *(const uint4*)(p + 16);
  return u.v;
}

// A fragment from an LDS-resident 16xKC row-major tile.
static __device__ __forceinline__ v16bf lds_a_frag(const u16* tile, int lane, int k0) {
  const int m  = lane & 15;
  const int kk = k0 + ((lane >> 4) << 3);
  const u16* p = tile + m * KC + kk;
  V16U u;
  u.q[0] = *(const uint4*)p;
  u.q[1] = *(const uint4*)(p + 16);
  return u.v;
}

static __device__ __forceinline__ v8f wmma_bf16(v16bf a, v16bf b, v8f c) {
  return __builtin_amdgcn_wmma_f32_16x16x32_bf16(false, a, false, b, (short)0, c,
                                                 false, false);
}

// CDNA5 async copy: 16 bytes global -> LDS, tracked by ASYNCcnt (ISA 15.18.3).
static __device__ __forceinline__ void async_copy_b128(unsigned lds_off,
                                                       const u16* g) {
  unsigned long long ga = (unsigned long long)(uintptr_t)g;
  asm volatile("global_load_async_to_lds_b128 %0, %1, off"
               :: "v"(lds_off), "v"(ga)
               : "memory");
}
static __device__ __forceinline__ void wait_async_le(int n) {
  if (n == 0) asm volatile("s_wait_asynccnt 0x0" ::: "memory");
  else        asm volatile("s_wait_asynccnt 0x1" ::: "memory");
}

static __device__ __forceinline__ float sigmoidf(float x) {
  return 1.0f / (1.0f + __expf(-x));
}

// ---------- conversion kernels ----------

__global__ __launch_bounds__(256) void concat_convert(const float* __restrict__ Hin,
                                                      const float* __restrict__ Ce,
                                                      u16* __restrict__ X) {
  int i = blockIdx.x * blockDim.x + threadIdx.x;          // over (Bsz*Tsz*Hd)/4
  int e = i << 2;
  int row = e >> 9;
  int col = e & 511;
  float4 h = *(const float4*)(Hin + e);
  float4 c = *(const float4*)(Ce + e);
  u16* pr = X + ((long)row << 10) + col;
  pr[0] = f2bf(h.x); pr[1] = f2bf(h.y); pr[2] = f2bf(h.z); pr[3] = f2bf(h.w);
  u16* pc = pr + Hd;
  pc[0] = f2bf(c.x); pc[1] = f2bf(c.y); pc[2] = f2bf(c.z); pc[3] = f2bf(c.w);
}

__global__ __launch_bounds__(256) void convert_bf16(const float* __restrict__ src,
                                                    u16* __restrict__ dst, int n4) {
  int i = blockIdx.x * blockDim.x + threadIdx.x;
  if (i >= n4) return;
  float4 v = *(const float4*)(src + (i << 2));
  u16* p = dst + (i << 2);
  p[0] = f2bf(v.x); p[1] = f2bf(v.y); p[2] = f2bf(v.z); p[3] = f2bf(v.w);
}

// ---------- C = A * W^T + bias  (A:[M,K] bf16 rm, W:[N,K] bf16 rm) ----------
// Block = 8 waves sharing one 16-row A tile (LDS, async double-buffered);
// each wave computes 16x64 with 4 WMMA accumulators. Two k-steps are kept in
// flight with distinct registers so B loads overlap WMMA execution.
// grid = (M/16, N/512). K must be a multiple of KC.

__global__ __launch_bounds__(256, 1) void gemm_bf16_nt(const u16* __restrict__ A,
                                                       const u16* __restrict__ W,
                                                       const float* __restrict__ bias,
                                                       float* __restrict__ C,
                                                       int K, int N) {
  __shared__ u16 smem[2][16 * KC];
  const int tid  = threadIdx.x;
  const int lane = tid & 31;
  const int wv   = tid >> 5;
  const int m0   = blockIdx.x << 4;
  const int n0   = blockIdx.y * 512 + wv * 64;
  const int ln   = lane & 15;

  // this thread's 16-byte slice of every A chunk: row tid>>4, cols (tid&15)*8..+7
  const u16* asrc = A + (long)(m0 + (tid >> 4)) * K + ((tid & 15) << 3);
  const unsigned lds0 = (unsigned)(uintptr_t)(void*)&smem[0][0] + (unsigned)tid * 16u;
  const unsigned lds1 = (unsigned)(uintptr_t)(void*)&smem[1][0] + (unsigned)tid * 16u;

  v8f acc[4];
#pragma unroll
  for (int j = 0; j < 4; ++j) {
    float b = bias[n0 + j * 16 + ln];
#pragma unroll
    for (int e = 0; e < 8; ++e) acc[j][e] = b;
  }

  const int nch = K / KC;
  async_copy_b128(lds0, asrc);                         // chunk 0
  for (int c = 0; c < nch; ++c) {
    if (c + 1 < nch) {
      async_copy_b128((c & 1) ? lds0 : lds1, asrc + (c + 1) * KC);
      wait_async_le(1);                                // chunk c complete (in-order)
    } else {
      wait_async_le(0);
    }
    __syncthreads();                                   // whole tile visible

    const u16* at = &smem[c & 1][0];
    const int  kb = c * KC;
#pragma unroll
    for (int p = 0; p < KC / 64; ++p) {                // two k-steps per pair
      const int k0 = p * 64;
      v16bf a0 = lds_a_frag(at, lane, k0);
      v16bf b00 = load_b_frag(W + (long)(n0 +  0) * K, K, lane, kb + k0);
      v16bf b01 = load_b_frag(W + (long)(n0 + 16) * K, K, lane, kb + k0);
      v16bf b02 = load_b_frag(W + (long)(n0 + 32) * K, K, lane, kb + k0);
      v16bf b03 = load_b_frag(W + (long)(n0 + 48) * K, K, lane, kb + k0);
      v16bf a1 = lds_a_frag(at, lane, k0 + 32);
      v16bf b10 = load_b_frag(W + (long)(n0 +  0) * K, K, lane, kb + k0 + 32);
      v16bf b11 = load_b_frag(W + (long)(n0 + 16) * K, K, lane, kb + k0 + 32);
      v16bf b12 = load_b_frag(W + (long)(n0 + 32) * K, K, lane, kb + k0 + 32);
      v16bf b13 = load_b_frag(W + (long)(n0 + 48) * K, K, lane, kb + k0 + 32);
      acc[0] = wmma_bf16(a0, b00, acc[0]);
      acc[1] = wmma_bf16(a0, b01, acc[1]);
      acc[2] = wmma_bf16(a0, b02, acc[2]);
      acc[3] = wmma_bf16(a0, b03, acc[3]);
      acc[0] = wmma_bf16(a1, b10, acc[0]);
      acc[1] = wmma_bf16(a1, b11, acc[1]);
      acc[2] = wmma_bf16(a1, b12, acc[2]);
      acc[3] = wmma_bf16(a1, b13, acc[3]);
    }
    __syncthreads();                                   // tile reads done before reuse
  }

  const int rbase = m0 + ((lane >> 4) << 3);
#pragma unroll
  for (int j = 0; j < 4; ++j)
#pragma unroll
    for (int r = 0; r < 8; ++r)
      C[(long)(rbase + r) * N + n0 + j * 16 + ln] = acc[j][r];
}

// ---------- persistent GRU scan ----------

static __device__ __forceinline__ void grid_barrier(unsigned* counter, unsigned* gen,
                                                    unsigned nblocks) {
  __threadfence();                       // release: publish this thread's stores
  __syncthreads();
  if (threadIdx.x == 0) {
    unsigned g = __atomic_load_n(gen, __ATOMIC_RELAXED);
    unsigned prev = __atomic_fetch_add(counter, 1u, __ATOMIC_ACQ_REL);
    if (prev + 1 == nblocks) {
      __atomic_store_n(counter, 0u, __ATOMIC_RELAXED);
      __atomic_fetch_add(gen, 1u, __ATOMIC_ACQ_REL);
    } else {
      while (__atomic_load_n(gen, __ATOMIC_ACQUIRE) == g)
        __builtin_amdgcn_s_sleep(2);
    }
  }
  __syncthreads();
  __builtin_amdgcn_fence(__ATOMIC_ACQUIRE, "agent");   // invalidate stale near caches
}

// 64 blocks x 128 threads = 256 waves (1 wave/SIMD -> huge register budget).
// Wave (mi,hi) owns batch rows mi*16..+15 and hidden cols hi*16..+15.
// All 48 recurrent-weight WMMA fragments (16 k-steps x 3 gates) are preloaded
// into registers once; the 256-step loop does no weight loads at all. The
// wave's own h tile is carried in registers across steps; only the cross-wave
// bf16 h exchange goes through memory (one grid barrier per timestep).
__global__ __launch_bounds__(128, 1) void gru_scan(const float* __restrict__ xg,
                                                   const u16*  __restrict__ whh,
                                                   const float* __restrict__ b_hh,
                                                   u16* __restrict__ hbuf0,
                                                   u16* __restrict__ hbuf1,
                                                   float* __restrict__ gout,
                                                   u16*  __restrict__ goutbf,
                                                   unsigned* counter, unsigned* gen) {
  const int lane = threadIdx.x & 31;
  const int wave = (blockIdx.x * blockDim.x + threadIdx.x) >> 5;   // 0..255
  const int mi = wave >> 5;          // 0..7
  const int hi = wave & 31;          // 0..31
  const int m0 = mi << 4;
  const int h0 = hi << 4;
  const int ln = lane & 15;
  const int lh = lane >> 4;

  // zero initial hidden state (buffer 0)
#pragma unroll
  for (int r = 0; r < 8; ++r)
    hbuf0[(m0 + r + lh * 8) * Hd + h0 + ln] = (u16)0;

  const float bh_r = b_hh[h0 + ln];
  const float bh_z = b_hh[Hd + h0 + ln];
  const float bh_n = b_hh[2 * Hd + h0 + ln];
  const u16* wr = whh + (long)(h0) * Hd;
  const u16* wz = whh + (long)(Hd + h0) * Hd;
  const u16* wn = whh + (long)(2 * Hd + h0) * Hd;

  // preload ALL recurrent weight fragments into registers (48 x v16bf)
  v16bf wc[48];
#pragma unroll
  for (int ks = 0; ks < 16; ++ks) {
    wc[3 * ks + 0] = load_b_frag(wr, Hd, lane, ks * 32);
    wc[3 * ks + 1] = load_b_frag(wz, Hd, lane, ks * 32);
    wc[3 * ks + 2] = load_b_frag(wn, Hd, lane, ks * 32);
  }

  float hreg[8];                       // this wave's own h tile (C layout)
#pragma unroll
  for (int r = 0; r < 8; ++r) hreg[r] = 0.0f;

  grid_barrier(counter, gen, gridDim.x);

  for (int t = 0; t < Tsz; ++t) {
    const u16* hcur  = (t & 1) ? hbuf1 : hbuf0;
    u16*       hnext = (t & 1) ? hbuf0 : hbuf1;

    v8f ar = {}, az = {}, an = {};
    const u16* Arow = hcur + (long)m0 * Hd;
#pragma unroll
    for (int ks = 0; ks < 16; ++ks) {
      v16bf a = load_a_frag(Arow, Hd, lane, ks * 32);
      ar = wmma_bf16(a, wc[3 * ks + 0], ar);
      az = wmma_bf16(a, wc[3 * ks + 1], az);
      an = wmma_bf16(a, wc[3 * ks + 2], an);
    }

#pragma unroll
    for (int r = 0; r < 8; ++r) {
      const int row = m0 + r + lh * 8;              // batch index
      const int col = h0 + ln;                      // hidden index
      const long xb = (long)(row * Tsz + t) * G3;
      float xr = xg[xb + col];
      float xz = xg[xb + Hd + col];
      float xn = xg[xb + 2 * Hd + col];
      float rg = sigmoidf(xr + ar[r] + bh_r);
      float zg = sigmoidf(xz + az[r] + bh_z);
      float ng = tanhf(xn + rg * (an[r] + bh_n));
      float hnew = (1.0f - zg) * ng + zg * hreg[r];
      hreg[r] = hnew;
      const long ob = (long)(row * Tsz + t) * Hd + col;
      gout[ob] = hnew;
      u16 hb = f2bf(hnew);
      goutbf[ob] = hb;
      hnext[row * Hd + col] = hb;
    }
    grid_barrier(counter, gen, gridDim.x);
  }
}

// ---------- fused LayerNorm + final projection, one wave per row ----------

__global__ __launch_bounds__(256) void ln_fc_kernel(const float* __restrict__ mid,
                                                    const float* __restrict__ lng,
                                                    const float* __restrict__ lnb,
                                                    const float* __restrict__ fw,
                                                    const float* __restrict__ fb,
                                                    float* __restrict__ out) {
  const int lane = threadIdx.x & 31;
  const int row  = (blockIdx.x << 3) + (threadIdx.x >> 5);
  const float* p = mid + (long)row * Hd + lane * 16;
  float x[16];
  float s = 0.0f;
#pragma unroll
  for (int i = 0; i < 4; ++i) {
    float4 v = *(const float4*)(p + i * 4);
    x[4 * i] = v.x; x[4 * i + 1] = v.y; x[4 * i + 2] = v.z; x[4 * i + 3] = v.w;
    s += v.x + v.y + v.z + v.w;
  }
#pragma unroll
  for (int o = 16; o; o >>= 1) s += __shfl_xor(s, o, 32);
  const float mu = s * (1.0f / 512.0f);
  float vs = 0.0f;
#pragma unroll
  for (int i = 0; i < 16; ++i) { float d = x[i] - mu; vs += d * d; }
#pragma unroll
  for (int o = 16; o; o >>= 1) vs += __shfl_xor(vs, o, 32);
  const float inv = rsqrtf(vs * (1.0f / 512.0f) + 1e-5f);
  float acc = 0.0f;
  const int c0 = lane * 16;
#pragma unroll
  for (int i = 0; i < 16; ++i) {
    int c = c0 + i;
    acc += ((x[i] - mu) * inv * lng[c] + lnb[c]) * fw[c];
  }
#pragma unroll
  for (int o = 16; o; o >>= 1) acc += __shfl_xor(acc, o, 32);
  if (lane == 0) out[row] = acc + fb[0];
}

// ---------- host ----------

extern "C" void kernel_launch(void* const* d_in, const int* in_sizes, int n_in,
                              void* d_out, int out_size, void* d_ws, size_t ws_size,
                              hipStream_t stream) {
  const float* Hin    = (const float*)d_in[0];
  const float* Cemb   = (const float*)d_in[1];
  const float* w_ih   = (const float*)d_in[2];
  const float* w_hh   = (const float*)d_in[3];
  const float* b_ih   = (const float*)d_in[4];
  const float* b_hh   = (const float*)d_in[5];
  const float* fcm_w  = (const float*)d_in[6];
  const float* fcm_b  = (const float*)d_in[7];
  const float* ln_g   = (const float*)d_in[8];
  const float* ln_b   = (const float*)d_in[9];
  const float* fc_w   = (const float*)d_in[10];
  const float* fc_b   = (const float*)d_in[11];
  (void)in_sizes; (void)n_in; (void)out_size; (void)ws_size;

  char* ws = (char*)d_ws;
  size_t off = 0;
  auto take = [&](size_t bytes) {
    char* p = ws + off;
    off = (off + bytes + 255) & ~(size_t)255;
    return p;
  };
  const long M = (long)Bsz * Tsz;                         // 32768
  u16*   Xbf    = (u16*)  take(M * Din * 2);              // concat input, bf16
  u16*   wihbf  = (u16*)  take((long)G3 * Din * 2);
  u16*   whhbf  = (u16*)  take((long)G3 * Hd * 2);
  u16*   fmbf   = (u16*)  take((long)Hd * Hd * 2);
  float* xg     = (float*)take(M * G3 * 4);
  float* gout   = (float*)take(M * Hd * 4);
  u16*   goutbf = (u16*)  take(M * Hd * 2);
  u16*   hbuf0  = (u16*)  take((long)Bsz * Hd * 2);
  u16*   hbuf1  = (u16*)  take((long)Bsz * Hd * 2);
  float* mid    = (float*)take(M * Hd * 4);
  unsigned* barrier = (unsigned*)take(2 * sizeof(unsigned));

  hipMemsetAsync(barrier, 0, 2 * sizeof(unsigned), stream);

  // 1) conversions
  {
    int n4 = (int)(M * Hd / 4);
    concat_convert<<<n4 / 256, 256, 0, stream>>>(Hin, Cemb, Xbf);
    int a = (G3 * Din) / 4;
    convert_bf16<<<(a + 255) / 256, 256, 0, stream>>>(w_ih, wihbf, a);
    int b = (G3 * Hd) / 4;
    convert_bf16<<<(b + 255) / 256, 256, 0, stream>>>(w_hh, whhbf, b);
    int c = (Hd * Hd) / 4;
    convert_bf16<<<(c + 255) / 256, 256, 0, stream>>>(fcm_w, fmbf, c);
  }

  // 2) xg = X @ w_ih^T + b_ih   [32768 x 1536]
  gemm_bf16_nt<<<dim3((unsigned)(M / 16), G3 / 512), 256, 0, stream>>>(
      Xbf, wihbf, b_ih, xg, Din, G3);

  // 3) persistent GRU scan (one grid barrier per timestep)
  gru_scan<<<64, 128, 0, stream>>>(xg, whhbf, b_hh, hbuf0, hbuf1, gout, goutbf,
                                   barrier, barrier + 1);

  // 4) mid = gru_out @ fc_mid_w^T + fc_mid_b   [32768 x 512]
  gemm_bf16_nt<<<dim3((unsigned)(M / 16), 1), 256, 0, stream>>>(
      goutbf, fmbf, fcm_b, mid, Hd, Hd);

  // 5) LayerNorm + final projection -> logits [32768]
  ln_fc_kernel<<<(unsigned)(M / 8), 256, 0, stream>>>(mid, ln_g, ln_b, fc_w, fc_b,
                                                      (float*)d_out);
}
